// SinkhornSelfAttention_87033217286671
// MI455X (gfx1250) — matrix operand order repaired and checked by reading, not compiled, not run on hardware
//
#include <hip/hip_runtime.h>
#include <hip/hip_bf16.h>

// ---------------------------------------------------------------------------
// Sinkhorn self-attention, CDNA5 (gfx1250) WMMA implementation.
// B=4, T=8192, DIM=512, HEADS=8, BUCKETS=128, DH=64, BSZ=64
// ---------------------------------------------------------------------------

typedef __attribute__((ext_vector_type(16))) __bf16 v16bf;
typedef __attribute__((ext_vector_type(8)))  float  v8f;

#define CB  4
#define CT  8192
#define CDIM 512
#define CH  8
#define CDH 64
#define CNB 128
#define CBSZ 64
#define CBH (CB*CH)          // 32
#define SCALE_D 0.044194173824159216f   // 512^-0.5

// ---- optional gfx1250 async global->LDS copy path (ASYNCcnt) --------------
// Builtin presence is used only as a feature probe (its pointer params use
// HIP's unspellable cuda_device/cuda_shared LangAS); the actual copy is
// emitted via inline asm per the VGLOBAL async encoding:
//   global_load_async_to_lds_b128 vDst(LDS byte addr), vAddr(64b global), off
#define HAVE_ASYNC_LDS 0
#if defined(__has_builtin)
#if __has_builtin(__builtin_amdgcn_global_load_async_to_lds_b128)
#undef HAVE_ASYNC_LDS
#define HAVE_ASYNC_LDS 1
#endif
#endif

// Per-lane 16-byte global->LDS copy. Async (ASYNCcnt-tracked, no VGPR data
// round-trip) when the gfx1250 path is available; otherwise b128 load +
// ds_store_b128.
__device__ inline void copy_b128_to_lds(const void* g, void* lds) {
#if HAVE_ASYNC_LDS
  unsigned ldsAddr = (unsigned)(size_t)lds;            // LDS byte offset
  unsigned long long gAddr = (unsigned long long)(size_t)g;
  asm volatile("global_load_async_to_lds_b128 %0, %1, off"
               :
               : "v"(ldsAddr), "v"(gAddr)
               : "memory");
#else
  *(uint4*)lds = *(const uint4*)g;
#endif
}

__device__ inline void async_copies_wait() {
#if HAVE_ASYNC_LDS
#if defined(__has_builtin) && __has_builtin(__builtin_amdgcn_s_wait_asynccnt)
  __builtin_amdgcn_s_wait_asynccnt(0);
#else
  asm volatile("s_wait_asynccnt 0x0" ::: "memory");
#endif
#endif
}

__device__ inline float bf2f(__bf16 h) { return (float)h; }
__device__ inline __bf16 f2bf(float f) { return (__bf16)f; }

__device__ inline v8f wmma_bf16(v16bf a, v16bf b, v8f c) {
  return __builtin_amdgcn_wmma_f32_16x16x32_bf16(false, a, false, b, (short)0, c,
                                                 false, false);
}

// A fragment (16x32, 16-bit): lanes 0-15 -> M=lane, K octets {0..7,16..23};
// lanes 16-31 -> M=lane-16, K octets {8..15,24..31}.  `base` points at K=0 of
// the current 32-wide K slab; `stride` is the LDS row stride in halfs.
__device__ inline v16bf frag_A(const __bf16* base, int stride, int mBase, int lane) {
  int m    = mBase + (lane & 15);
  int koff = (lane >> 4) * 8;
  const __bf16* p = base + m * stride + koff;
  v16bf f;
#pragma unroll
  for (int e = 0; e < 8; ++e) f[e] = p[e];
#pragma unroll
  for (int e = 0; e < 8; ++e) f[e + 8] = p[e + 16];
  return f;
}

// B fragment (32x16, 16-bit) from a K-contiguous ("transposed") LDS tile:
// lanes 0-15 -> N=lane, K octets {0..7,16..23}; lanes 16-31 -> N=lane-16, K octets {8..15,24..31}.
__device__ inline v16bf frag_B(const __bf16* baseT, int strideK, int nBase, int lane) {
  int n  = nBase + (lane & 15);
  int kb = (lane >> 4) * 8;
  const __bf16* p = baseT + n * strideK + kb;
  v16bf f;
#pragma unroll
  for (int e = 0; e < 8; ++e) f[e] = p[e];
#pragma unroll
  for (int e = 0; e < 8; ++e) f[e + 8] = p[e + 16];
  return f;
}

// ---------------------------------------------------------------------------
// f32 -> bf16 conversion
// ---------------------------------------------------------------------------
__global__ void cvt_bf16_kernel(const float* __restrict__ src,
                                __bf16* __restrict__ dst, long n) {
  long i = (long)blockIdx.x * blockDim.x + threadIdx.x;
  long stride = (long)gridDim.x * blockDim.x;
  for (; i < n; i += stride) dst[i] = f2bf(src[i]);
}

// ---------------------------------------------------------------------------
// Staged bf16 WMMA GEMM:  C(MxN) = A(MxK) * B(KxN), A/B row-major bf16.
// Block tile 64x128, 256 threads (8 waves); wave w owns m-subtile (w&3)*16 and
// n-half (w>>2)*64 (4 wmma tiles).  K-step 32.
// mode 0: scatter bf16 into bucketed qkv layout [which][bh][t][c]
// mode 1: f32 out + bias (final projection)
// ---------------------------------------------------------------------------
__global__ __launch_bounds__(256) void gemm_bf16_kernel(
    const __bf16* __restrict__ A, const __bf16* __restrict__ B,
    int M, int N, int K, int mode,
    __bf16* __restrict__ bOut, float* __restrict__ fOut,
    const float* __restrict__ bias) {
  __shared__ __align__(16) __bf16 ldsA[64 * 32];    // row-major [m][k]
  __shared__ __align__(16) __bf16 ldsBt[128 * 32];  // transposed [n][k]

  const int tid  = threadIdx.x;
  const int lane = tid & 31;
  const int wave = tid >> 5;
  const int mBlk = blockIdx.x * 64;
  const int nBlk = blockIdx.y * 128;
  const int mSub  = (wave & 3) * 16;
  const int nHalf = (wave >> 2) * 64;

  v8f acc[4];
#pragma unroll
  for (int i = 0; i < 4; ++i) acc[i] = v8f{0.f, 0.f, 0.f, 0.f, 0.f, 0.f, 0.f, 0.f};

  const int kSteps = K >> 5;
  for (int ks = 0; ks < kSteps; ++ks) {
    const int k0 = ks * 32;
    // stage A: 64x32 halfs, one 16B lane-copy per thread (async when avail.)
    {
      int idx = tid * 8;
      int r = idx >> 5, c = idx & 31;
      copy_b128_to_lds(A + (size_t)(mBlk + r) * K + k0 + c, &ldsA[r * 32 + c]);
      if (ks + 1 < kSteps)
        __builtin_prefetch(A + (size_t)(mBlk + r) * K + k0 + 32 + c, 0, 0);
    }
    // stage B transposed: 32x128 halfs, two uint4 per thread, scatter [n][k]
#pragma unroll
    for (int it = 0; it < 2; ++it) {
      int idx = (tid + it * 256) * 8;
      int r = idx >> 7, c = idx & 127;
      uint4 v = *(const uint4*)(B + (size_t)(k0 + r) * N + nBlk + c);
      const __bf16* h = (const __bf16*)&v;
#pragma unroll
      for (int j = 0; j < 8; ++j) ldsBt[(c + j) * 32 + r] = h[j];
    }
    async_copies_wait();
    __syncthreads();

    v16bf aF = frag_A(ldsA, 32, mSub, lane);
#pragma unroll
    for (int nt = 0; nt < 4; ++nt) {
      v16bf bF = frag_B(ldsBt, 32, nHalf + nt * 16, lane);
      acc[nt] = wmma_bf16(aF, bF, acc[nt]);
    }
    __syncthreads();
  }

  // store: VGPR r -> M = r + 8*(lane>=16); N = lane&15
#pragma unroll
  for (int nt = 0; nt < 4; ++nt) {
#pragma unroll
    for (int r = 0; r < 8; ++r) {
      int m = mBlk + mSub + r + 8 * (lane >> 4);
      int n = nBlk + nHalf + nt * 16 + (lane & 15);
      float v = acc[nt][r];
      if (mode == 0) {
        int which = n >> 9;        // q/k/v
        int rem = n & 511;
        int h = rem >> 6, c = rem & 63;
        int b = m >> 13, t = m & 8191;
        size_t dst = ((size_t)which * CBH * CT +
                      (size_t)(b * CH + h) * CT + t) * CDH + c;
        bOut[dst] = f2bf(v);
      } else {
        fOut[(size_t)m * N + n] = v + bias[n];
      }
    }
  }
}

// ---------------------------------------------------------------------------
// sfeat[bh][u][0:64]  = sum_i q[bh][u*64+i][c]
// sfeat[bh][u][64:128]= sum_i k[bh][u*64+i][c]
// ---------------------------------------------------------------------------
__global__ __launch_bounds__(128) void bucket_sums_kernel(
    const __bf16* __restrict__ qb, const __bf16* __restrict__ kb,
    float* __restrict__ sfeat) {
  int blk = blockIdx.x;                 // bh*128 + u
  int u = blk & 127, bh = blk >> 7;
  int tid = threadIdx.x;                // 0..127
  const __bf16* base = (tid < 64) ? qb : kb;
  int c = tid & 63;
  float s = 0.f;
  size_t row0 = ((size_t)bh * CT + (size_t)u * CBSZ) * CDH + c;
  for (int i = 0; i < CBSZ; ++i) s += bf2f(base[row0 + (size_t)i * CDH]);
  sfeat[((size_t)blk) * 128 + tid] = s;
}

// ---------------------------------------------------------------------------
// sortnet: R = relu(sfeat @ w_sort[h]); softmax row; keep top-1:
// src[bh][u] = argmax, scale[bh][u] = max softmax value = 1/sum(exp(r-rmax))
// ---------------------------------------------------------------------------
__global__ __launch_bounds__(128) void sortnet_kernel(
    const float* __restrict__ sfeat, const float* __restrict__ w_sort,
    int* __restrict__ srcIdx, float* __restrict__ topScale) {
  __shared__ float sF[128];
  __shared__ float red[128];
  __shared__ int redI[128];
  int blk = blockIdx.x;                 // bh*128 + u
  int bh = blk >> 7;
  int h = bh & 7;
  int m = threadIdx.x;
  sF[m] = sfeat[(size_t)blk * 128 + m];
  __syncthreads();
  float r = 0.f;
  for (int f = 0; f < 128; ++f)
    r += sF[f] * w_sort[((size_t)h * 128 + f) * 128 + m];
  r = fmaxf(r, 0.f);
  // row max
  red[m] = r; __syncthreads();
  for (int s2 = 64; s2 > 0; s2 >>= 1) {
    if (m < s2) red[m] = fmaxf(red[m], red[m + s2]);
    __syncthreads();
  }
  float rmax = red[0]; __syncthreads();
  // exp-sum
  red[m] = __expf(r - rmax); __syncthreads();
  for (int s2 = 64; s2 > 0; s2 >>= 1) {
    if (m < s2) red[m] += red[m + s2];
    __syncthreads();
  }
  float esum = red[0]; __syncthreads();
  // argmax, first-index tie-break
  red[m] = r; redI[m] = m; __syncthreads();
  for (int s2 = 64; s2 > 0; s2 >>= 1) {
    if (m < s2) {
      float a = red[m], b2 = red[m + s2];
      if (b2 > a || (b2 == a && redI[m + s2] < redI[m])) {
        red[m] = b2; redI[m] = redI[m + s2];
      }
    }
    __syncthreads();
  }
  if (m == 0) { srcIdx[blk] = redI[0]; topScale[blk] = 1.f / esum; }
}

// ---------------------------------------------------------------------------
// attention: one block per half-bucket (32 rows of a 64-token bucket).
// K2/V2 = [scaled gather from bucket src ; local bucket]  (128 x 64)
// dots = Q @ K2^T * 512^-0.5 ; softmax ; out = P @ V2 -> bf16 ctx (b,t,dim)
// ---------------------------------------------------------------------------
__global__ __launch_bounds__(256) void attn_kernel(
    const __bf16* __restrict__ qb, const __bf16* __restrict__ kb,
    const __bf16* __restrict__ vb, const int* __restrict__ srcIdx,
    const float* __restrict__ topScale, __bf16* __restrict__ ctx) {
  __shared__ __align__(16) char smem[53248];
  __bf16* ldsQ  = (__bf16*)(smem);              // 32x64  row-major   4KB
  __bf16* ldsK  = (__bf16*)(smem + 4096);       // 128x64 row-major  16KB
  __bf16* ldsVt = (__bf16*)(smem + 20480);      // [c][j] 64x128     16KB
  float*  ldsS  = (float*)(smem + 36864);       // 32x128 f32        16KB
  __bf16* ldsP  = (__bf16*)(smem);              // 32x128 bf16, aliases Q/K

  const int tid  = threadIdx.x;
  const int lane = tid & 31;
  const int wave = tid >> 5;
  const int blk  = blockIdx.x;          // bh*256 + u*2 + half
  const int halfSel = blk & 1;
  const int u  = (blk >> 1) & 127;
  const int bh = blk >> 8;
  const int vsrc = srcIdx[bh * CNB + u];
  const float scl = topScale[bh * CNB + u];
  const int t0 = u * CBSZ + halfSel * 32;

  // ---- stage Q (32x64) : one 16B lane-copy per thread (async when avail.) --
  {
    int idx = tid * 8;
    int r = idx >> 6, c = idx & 63;
    copy_b128_to_lds(qb + ((size_t)bh * CT + t0 + r) * CDH + c,
                     &ldsQ[r * 64 + c]);
  }
  // ---- stage K2 (128x64) and V2^T : 4 uint4 per thread each ----
#pragma unroll
  for (int it = 0; it < 4; ++it) {
    int idx = (tid + it * 256) * 8;
    int j = idx >> 6, c = idx & 63;
    size_t srow = (j < 64)
        ? ((size_t)bh * CT + (size_t)vsrc * CBSZ + j) * CDH + c
        : ((size_t)bh * CT + (size_t)u * CBSZ + (j - 64)) * CDH + c;
    float s = (j < 64) ? scl : 1.f;
    // K2
    {
      uint4 v = *(const uint4*)(kb + srow);
      const __bf16* hh = (const __bf16*)&v;
      __align__(16) __bf16 tmp[8];
#pragma unroll
      for (int e = 0; e < 8; ++e) tmp[e] = f2bf(bf2f(hh[e]) * s);
      *(uint4*)(&ldsK[j * 64 + c]) = *(uint4*)tmp;
    }
    // V2 transposed scatter
    {
      uint4 v = *(const uint4*)(vb + srow);
      const __bf16* hh = (const __bf16*)&v;
#pragma unroll
      for (int e = 0; e < 8; ++e) ldsVt[(c + e) * 128 + j] = f2bf(bf2f(hh[e]) * s);
    }
  }
  async_copies_wait();
  __syncthreads();

  // ---- dots = Q @ K2^T : 2 m-tiles x 8 n-tiles, 2 tiles/wave, K=64 ----
#pragma unroll
  for (int i = 0; i < 2; ++i) {
    int tile = wave * 2 + i;
    int mT = tile >> 3, nT = tile & 7;
    v8f acc = v8f{0.f, 0.f, 0.f, 0.f, 0.f, 0.f, 0.f, 0.f};
#pragma unroll
    for (int ks = 0; ks < 2; ++ks) {
      v16bf aF = frag_A(ldsQ + ks * 32, 64, mT * 16, lane);
      v16bf bF = frag_B(ldsK + ks * 32, 64, nT * 16, lane);
      acc = wmma_bf16(aF, bF, acc);
    }
#pragma unroll
    for (int r = 0; r < 8; ++r) {
      int mloc = mT * 16 + r + 8 * (lane >> 4);
      int nloc = nT * 16 + (lane & 15);
      ldsS[mloc * 128 + nloc] = acc[r] * SCALE_D;
    }
  }
  __syncthreads();

  // ---- softmax (one thread per row) -> bf16 probs in ldsP ----
  if (tid < 32) {
    float* row = ldsS + tid * 128;
    float mx = row[0];
    for (int j = 1; j < 128; ++j) mx = fmaxf(mx, row[j]);
    float sum = 0.f;
    for (int j = 0; j < 128; ++j) sum += __expf(row[j] - mx);
    float inv = 1.f / sum;
    for (int j = 0; j < 128; ++j)
      ldsP[tid * 128 + j] = f2bf(__expf(row[j] - mx) * inv);
  }
  __syncthreads();

  // ---- out = P @ V2 : 2 m-tiles x 4 n-tiles, 1 tile/wave, K=128 ----
  {
    int mT = wave >> 2, nT = wave & 3;
    v8f acc = v8f{0.f, 0.f, 0.f, 0.f, 0.f, 0.f, 0.f, 0.f};
#pragma unroll
    for (int ks = 0; ks < 4; ++ks) {
      v16bf aF = frag_A(ldsP + ks * 32, 128, mT * 16, lane);
      v16bf bF = frag_B(ldsVt + ks * 32, 128, nT * 16, lane);
      acc = wmma_bf16(aF, bF, acc);
    }
    int b = bh >> 3, h = bh & 7;
#pragma unroll
    for (int r = 0; r < 8; ++r) {
      int mloc = mT * 16 + r + 8 * (lane >> 4);
      int t = t0 + mloc;
      int c = nT * 16 + (lane & 15);
      ctx[((size_t)b * CT + t) * CDIM + h * CDH + c] = f2bf(acc[r]);
    }
  }
}

// ---------------------------------------------------------------------------
extern "C" void kernel_launch(void* const* d_in, const int* in_sizes, int n_in,
                              void* d_out, int out_size, void* d_ws, size_t ws_size,
                              hipStream_t stream) {
  const float* x      = (const float*)d_in[0];
  const float* w_qkv  = (const float*)d_in[1];
  const float* w_sort = (const float*)d_in[2];
  const float* w_out  = (const float*)d_in[3];
  const float* b_out  = (const float*)d_in[4];
  float* out = (float*)d_out;

  char* ws = (char*)d_ws;
  __bf16* xb    = (__bf16*)ws;  ws += (size_t)CB * CT * CDIM * 2;        // 32 MB
  __bf16* wqkvb = (__bf16*)ws;  ws += (size_t)CDIM * (3 * CDIM) * 2;     // 1.5 MB
  __bf16* woutb = (__bf16*)ws;  ws += (size_t)CDIM * CDIM * 2;           // 0.5 MB
  __bf16* qkvb  = (__bf16*)ws;  ws += (size_t)3 * CBH * CT * CDH * 2;    // 96 MB
  float*  sfeat = (float*)ws;   ws += (size_t)CBH * CNB * 128 * 4;       // 2 MB
  int*    srcI  = (int*)ws;     ws += (size_t)CBH * CNB * 4;
  float*  topS  = (float*)ws;   ws += (size_t)CBH * CNB * 4;
  __bf16* ctx   = (__bf16*)ws;  ws += (size_t)CB * CT * CDIM * 2;        // 32 MB

  const __bf16* qb = qkvb;
  const __bf16* kb = qkvb + (size_t)CBH * CT * CDH;
  const __bf16* vb = qkvb + (size_t)2 * CBH * CT * CDH;

  // 1) convert inputs to bf16
  cvt_bf16_kernel<<<4096, 256, 0, stream>>>(x, xb, (long)CB * CT * CDIM);
  cvt_bf16_kernel<<<768, 256, 0, stream>>>(w_qkv, wqkvb, (long)CDIM * 3 * CDIM);
  cvt_bf16_kernel<<<256, 256, 0, stream>>>(w_out, woutb, (long)CDIM * CDIM);

  // 2) qkv = x @ w_qkv -> bucketed bf16 q/k/v
  gemm_bf16_kernel<<<dim3((CB * CT) / 64, (3 * CDIM) / 128), 256, 0, stream>>>(
      xb, wqkvb, CB * CT, 3 * CDIM, CDIM, /*mode=*/0, qkvb, nullptr, nullptr);

  // 3) bucket feature sums
  bucket_sums_kernel<<<CBH * CNB, 128, 0, stream>>>(qb, kb, sfeat);

  // 4) sortnet -> per-bucket gather index + scale
  sortnet_kernel<<<CBH * CNB, 128, 0, stream>>>(sfeat, w_sort, srcI, topS);

  // 5) bucketed attention -> bf16 context (b,t,dim)
  attn_kernel<<<CBH * CNB * 2, 256, 0, stream>>>(qb, kb, vb, srcI, topS, ctx);

  // 6) out = ctx @ w_out + b_out (f32)
  gemm_bf16_kernel<<<dim3((CB * CT) / 64, CDIM / 128), 256, 0, stream>>>(
      ctx, woutb, CB * CT, CDIM, CDIM, /*mode=*/1, nullptr, out, b_out);
}
